// Net_5334349382151
// MI455X (gfx1250) — matrix operand order repaired and checked by reading, not compiled, not run on hardware
//
#include <hip/hip_runtime.h>
#include <hip/hip_bf16.h>

// ---------------------------------------------------------------------------
// GNN ranking model on MI455X (gfx1250).
// Heavy GEMMs ([64000 x 288/256] @ [K x 256]) run on v_wmma_f32_16x16x32_bf16
// (wave32, 16x16 tiles, f32 accumulate). Weights are repacked into exact
// WMMA B-fragment order so each lane streams 32B contiguous per fragment.
// The K-loop is double-buffered (fully unrolled via template) so each WMMA
// burst waits only on loads issued one iteration earlier.
// ---------------------------------------------------------------------------

typedef __attribute__((ext_vector_type(16))) __bf16 v16bf;
typedef __attribute__((ext_vector_type(8)))  float  v8f;
typedef __attribute__((ext_vector_type(4)))  unsigned int u32x4;

#define BB   2
#define NN   2000
#define CCFG 32
#define NCC  512
#define EE   6000
#define OPF  140
#define CFGF 18
#define EMBD 128
#define HIDD 256

__device__ __forceinline__ float gelu_f(float x) {
  return 0.5f * x * (1.0f + erff(x * 0.7071067811865476f));
}
__device__ __forceinline__ float elu_f(float x) {
  return x > 0.0f ? x : expm1f(x);
}

union bfrag { v16bf v; u32x4 u[2]; };

// Fully-unrolled, double-buffered K loop. KT = Kpad/32 (8 or 9).
template <int KT>
__device__ __forceinline__ void gemm_kloop(const __hip_bfloat16* __restrict__ arow,
                                           const u32x4* __restrict__ bbase,
                                           size_t bstride, v8f acc[4])
{
  bfrag af[2];
  bfrag bf[2][4];
  af[0].u[0] = *(const u32x4*)(arow);
  af[0].u[1] = *(const u32x4*)(arow + 16);
#pragma unroll
  for (int j = 0; j < 4; ++j) {
    bf[0][j].u[0] = bbase[j * 64];
    bf[0][j].u[1] = bbase[j * 64 + 1];
  }
#pragma unroll
  for (int kt = 0; kt < KT; ++kt) {
    const int cur = kt & 1;
    const int nxt = cur ^ 1;
    if (kt + 1 < KT) {  // prefetch next K-slice while current WMMAs issue
      af[nxt].u[0] = *(const u32x4*)(arow + (kt + 1) * 32);
      af[nxt].u[1] = *(const u32x4*)(arow + (kt + 1) * 32 + 16);
      const u32x4* bp = bbase + (size_t)(kt + 1) * bstride;
#pragma unroll
      for (int j = 0; j < 4; ++j) {
        bf[nxt][j].u[0] = bp[j * 64];
        bf[nxt][j].u[1] = bp[j * 64 + 1];
      }
    }
#pragma unroll
    for (int j = 0; j < 4; ++j) {
      acc[j] = __builtin_amdgcn_wmma_f32_16x16x32_bf16(
          false, af[cur].v, false, bf[cur][j].v, (short)0, acc[j], false, false);
    }
  }
}

// --------------------------------------------------------------------------
// WMMA GEMM: Y[M,Nout] = A[M,Kpad](bf16) * Bpacked + bias, f32 accumulate.
// Block = 8 waves; wave computes a 16x64 tile. Kpad in {256, 288},
// M % 128 == 0, Nout % 64 == 0.
// Bpacked layout: [(kt*ntiles + ntile)*32 + lane][e] bf16, with
//   n = ntile*16 + (lane&15), k = kt*32 + 8*(lane>>4) + (e<8 ? e : 8+e).
// A fragment (16-bit A 16x32): lane m = lane&15, g = lane>>4,
//   elems e<8 -> k = kt*32 + 8g + e ; e>=8 -> k = kt*32 + 8g + 8 + e.
// D layout: n = lane&15, row r holds m = r + 8g.
// --------------------------------------------------------------------------
__global__ __launch_bounds__(256) void gemm_bf16_wmma(
    const __hip_bfloat16* __restrict__ A, const __hip_bfloat16* __restrict__ Bp,
    const float* __restrict__ bias, float* __restrict__ Y,
    int M, int Kpad, int Nout)
{
  const int lane = threadIdx.x & 31;
  const int wave = threadIdx.x >> 5;
  const int rowTile = (blockIdx.x * 8 + wave) * 16;
  const int colBase = blockIdx.y * 64;
  const int ntiles = Nout >> 4;
  const int ktiles = Kpad >> 5;
  const int m = lane & 15;
  const int g = lane >> 4;
  const int nt0 = colBase >> 4;

  v8f acc[4];
#pragma unroll
  for (int j = 0; j < 4; ++j)
#pragma unroll
    for (int r = 0; r < 8; ++r) acc[j][r] = 0.0f;

  const __hip_bfloat16* arow = A + (size_t)(rowTile + m) * Kpad + 8 * g;
  const u32x4* bbase = (const u32x4*)Bp + ((size_t)nt0 * 32 + lane) * 2;
  const size_t bstride = (size_t)ntiles * 64;  // u32x4 units per K-tile

  if (ktiles == 9)      gemm_kloop<9>(arow, bbase, bstride, acc);
  else                  gemm_kloop<8>(arow, bbase, bstride, acc);

  const int n = lane & 15;
#pragma unroll
  for (int j = 0; j < 4; ++j) {
    int col = colBase + j * 16 + n;
    float b = bias ? bias[col] : 0.0f;
#pragma unroll
    for (int r = 0; r < 8; ++r) {
      Y[(size_t)(rowTile + 8 * g + r) * Nout + col] = acc[j][r] + b;
    }
  }
}

// Pack f32 weight [K, Nout] into WMMA B-fragment order (bf16), zero-pad k>=K.
__global__ void pack_weight_bf16(const float* __restrict__ W,
                                 __hip_bfloat16* __restrict__ out,
                                 int K, int Nout, int ktiles)
{
  int idx = blockIdx.x * blockDim.x + threadIdx.x;
  int ntiles = Nout >> 4;
  int total = ktiles * ntiles * 512;
  if (idx >= total) return;
  int e = idx & 15;
  int lane = (idx >> 4) & 31;
  int ntile = (idx >> 9) % ntiles;
  int kt = idx / (512 * ntiles);
  int nn = ntile * 16 + (lane & 15);
  int k = kt * 32 + 8 * (lane >> 4) + (e < 8 ? e : 8 + e);
  float v = (k < K) ? W[(size_t)k * Nout + nn] : 0.0f;
  out[idx] = __float2bfloat16(v);
}

// Embedding table with torch-style max_norm=1 renorm, output bf16 [120,128].
__global__ void renorm_embed(const float* __restrict__ emb,
                             __hip_bfloat16* __restrict__ out)
{
  int r = threadIdx.x;
  if (r >= 120) return;
  float ss = 0.0f;
  for (int i = 0; i < EMBD; ++i) { float v = emb[r * EMBD + i]; ss += v * v; }
  float nrm = sqrtf(ss);
  float sc = fminf(1.0f, 1.0f / (nrm + 1e-7f));
  for (int i = 0; i < EMBD; ++i)
    out[r * EMBD + i] = __float2bfloat16(emb[r * EMBD + i] * sc);
}

// Build early-MLP input A[C*N, 288] bf16: [opemb | nfz | zeros(cfg)].
__global__ void build_early_A(const float* __restrict__ nf,
                              const int* __restrict__ opc,
                              const __hip_bfloat16* __restrict__ embBf,
                              const float* __restrict__ fmean,
                              const float* __restrict__ fstd,
                              __hip_bfloat16* __restrict__ A)
{
  size_t idx = (size_t)blockIdx.x * blockDim.x + threadIdx.x;
  size_t total = (size_t)CCFG * NN * 288;
  if (idx >= total) return;
  int col = (int)(idx % 288);
  size_t row = idx / 288;
  int nd = (int)(row % NN);
  __hip_bfloat16 o;
  if (col < EMBD) {
    o = embBf[(size_t)opc[nd] * EMBD + col];
  } else if (col < EMBD + OPF) {
    int f = col - EMBD;
    float v = (nf[(size_t)nd * OPF + f] - fmean[f]) / (fstd[f] + 1e-4f);
    o = __float2bfloat16(v);
  } else {
    o = __float2bfloat16(0.0f);
  }
  A[idx] = o;
}

// Scatter per-config features into cfg columns of A (cids are unique).
__global__ void scatter_cfg(const float* __restrict__ cf,
                            const int* __restrict__ cids,
                            __hip_bfloat16* __restrict__ A)
{
  int idx = blockIdx.x * blockDim.x + threadIdx.x;
  if (idx >= CCFG * NCC * CFGF) return;
  int f = idx % CFGF;
  int i = (idx / CFGF) % NCC;
  int c = idx / (CFGF * NCC);
  int node = cids[i];
  A[((size_t)c * NN + node) * 288 + (EMBD + OPF) + f] = __float2bfloat16(cf[idx]);
}

// Partial column stats over the node axis: per (config, feature) sum & sumsq.
__global__ void col_stats(const float* __restrict__ Y, float* __restrict__ sumB,
                          float* __restrict__ sqB, int nsplit)
{
  int c = blockIdx.x;
  int f = threadIdx.x;        // F == HIDD == 256 == blockDim
  int part = blockIdx.y;
  int per = (NN + nsplit - 1) / nsplit;
  int n0 = part * per, n1 = n0 + per;
  if (n1 > NN) n1 = NN;
  float s = 0.0f, q = 0.0f;
  const float* base = Y + ((size_t)c * NN + n0) * HIDD + f;
  for (int n = n0; n < n1; ++n) { float v = *base; s += v; q += v * v; base += HIDD; }
  atomicAdd(&sumB[c * HIDD + f], s);
  atomicAdd(&sqB[c * HIDD + f], q);
}

// Normalize over nodes (eps 1e-5, biased var) + activation; emit f32/bf16.
__global__ void norm_apply(const float* __restrict__ Y,
                           const float* __restrict__ sumB,
                           const float* __restrict__ sqB,
                           float* __restrict__ outF32,
                           __hip_bfloat16* __restrict__ outBf,
                           int act /*0=gelu,1=elu*/)
{
  size_t idx = (size_t)blockIdx.x * blockDim.x + threadIdx.x;
  size_t total = (size_t)CCFG * NN * HIDD;
  if (idx >= total) return;
  int f = (int)(idx % HIDD);
  size_t row = idx / HIDD;
  int c = (int)(row / NN);
  const float invN = 1.0f / (float)NN;
  float m = sumB[c * HIDD + f] * invN;
  float var = sqB[c * HIDD + f] * invN - m * m;
  float r = rsqrtf(var + 1e-5f);
  float x = (Y[idx] - m) * r;
  float o = (act == 0) ? gelu_f(x) : elu_f(x);
  if (outF32) outF32[idx] = o;
  if (outBf)  outBf[idx] = __float2bfloat16(o);
}

// GIN edge aggregation: agg[c,dst,:] += h[c,src,:] for every edge.
__global__ void edge_agg(const float* __restrict__ H, const int* __restrict__ ei,
                         float* __restrict__ agg)
{
  int e = blockIdx.x;
  int c = blockIdx.y;
  int f = threadIdx.x;
  int s = ei[e];
  int d = ei[EE + e];
  atomicAdd(&agg[((size_t)c * NN + d) * HIDD + f],
            H[((size_t)c * NN + s) * HIDD + f]);
}

// (h + agg) -> bf16 matmul input.
__global__ void combine_bf(const float* __restrict__ H, const float* __restrict__ agg,
                           __hip_bfloat16* __restrict__ A)
{
  size_t idx = (size_t)blockIdx.x * blockDim.x + threadIdx.x;
  size_t total = (size_t)CCFG * NN * HIDD;
  if (idx >= total) return;
  A[idx] = __float2bfloat16(H[idx] + agg[idx]);
}

// pool[c,f] += mean over nodes of H[c,:,f]  (partial, atomic).
__global__ void pool_acc(const float* __restrict__ H, float* __restrict__ pool,
                         int nsplit)
{
  int c = blockIdx.x;
  int f = threadIdx.x;
  int part = blockIdx.y;
  int per = (NN + nsplit - 1) / nsplit;
  int n0 = part * per, n1 = n0 + per;
  if (n1 > NN) n1 = NN;
  float s = 0.0f;
  const float* base = H + ((size_t)c * NN + n0) * HIDD + f;
  for (int n = n0; n < n1; ++n) { s += *base; base += HIDD; }
  atomicAdd(&pool[c * HIDD + f], s * (1.0f / (float)NN));
}

// Small f32 GEMM for the late stage (32xK @ KxNout).
__global__ void small_gemm(const float* __restrict__ X, const float* __restrict__ W,
                           float* __restrict__ Z, int Mrows, int K, int Nout)
{
  int idx = blockIdx.x * blockDim.x + threadIdx.x;
  if (idx >= Mrows * Nout) return;
  int r = idx / Nout, n = idx % Nout;
  float s = 0.0f;
  for (int k = 0; k < K; ++k) s += X[(size_t)r * K + k] * W[(size_t)k * Nout + n];
  Z[idx] = s;
}

// Norm over the config axis (rows) + exact GELU. One block, F threads.
__global__ void normcfg_gelu(const float* __restrict__ Z, float* __restrict__ H,
                             int Crows, int F)
{
  int f = threadIdx.x;
  if (f >= F) return;
  float s = 0.0f, q = 0.0f;
  for (int c = 0; c < Crows; ++c) { float v = Z[c * F + f]; s += v; q += v * v; }
  float m = s / (float)Crows;
  float var = q / (float)Crows - m * m;
  float r = rsqrtf(var + 1e-5f);
  for (int c = 0; c < Crows; ++c) {
    float x = (Z[c * F + f] - m) * r;
    H[c * F + f] = gelu_f(x);
  }
}

__global__ void pred_k(const float* __restrict__ H2, const float* __restrict__ pw,
                       const float* __restrict__ pb, float* __restrict__ rt,
                       int K, int Crows)
{
  int c = threadIdx.x;
  if (c >= Crows) return;
  float s = pb[0];
  for (int k = 0; k < K; ++k) s += H2[c * K + k] * pw[k];
  rt[c] = s;
}

// Pairwise logistic ranking loss + OPA; also copies runtimes to d_out[0..63].
__global__ void loss_k(const float* __restrict__ rt, const float* __restrict__ tgt,
                       float* __restrict__ out)
{
  __shared__ float sl[256], so[256], sm[256];
  int t = threadIdx.x;
  float L = 0.0f, O = 0.0f, Mk = 0.0f;
  const int total = BB * CCFG * CCFG;
  for (int idx = t; idx < total; idx += blockDim.x) {
    int b = idx / (CCFG * CCFG);
    int rem = idx % (CCFG * CCFG);
    int i = rem / CCFG, j = rem % CCFG;
    float dp = rt[b * CCFG + i] - rt[b * CCFG + j];
    float dt = tgt[b * CCFG + i] - tgt[b * CCFG + j];
    if (dt > 0.0f) {
      Mk += 1.0f;
      float z = -dp;  // softplus(-dp), numerically stable
      L += fmaxf(z, 0.0f) + log1pf(expf(-fabsf(z)));
      if (dp > 0.0f) O += 1.0f;
    }
  }
  sl[t] = L; so[t] = O; sm[t] = Mk;
  __syncthreads();
  for (int s = 128; s > 0; s >>= 1) {
    if (t < s) { sl[t] += sl[t + s]; so[t] += so[t + s]; sm[t] += sm[t + s]; }
    __syncthreads();
  }
  if (t < BB * CCFG) out[t] = rt[t];
  if (t == 0) {
    float d = fmaxf(sm[0], 1.0f);
    out[BB * CCFG]     = sl[0] / d;
    out[BB * CCFG + 1] = so[0] / d;
  }
}

// ---------------------------------------------------------------------------
extern "C" void kernel_launch(void* const* d_in, const int* in_sizes, int n_in,
                              void* d_out, int out_size, void* d_ws, size_t ws_size,
                              hipStream_t stream)
{
  const float* node_feat        = (const float*)d_in[0];
  const int*   node_opcode      = (const int*)  d_in[1];
  const float* node_config_feat = (const float*)d_in[2];
  const int*   node_config_ids  = (const int*)  d_in[3];
  const int*   edge_index       = (const int*)  d_in[4];
  const float* config_runtime   = (const float*)d_in[5];

  // Identify param leaves by element count; the ordinal sequences are
  // identical for insertion-order and sorted-key pytree flattening.
  const float *feat_mean = nullptr, *feat_std = nullptr, *embed = nullptr;
  const float *early_w = nullptr, *late_w1 = nullptr, *late_w2 = nullptr;
  const float *pred_w = nullptr, *pred_b = nullptr;
  const float *gin_w[8] = {}, *gin_b[8] = {};
  int c65536 = 0, c256 = 0, c140 = 0;
  for (int i = 6; i < n_in; ++i) {
    const float* p = (const float*)d_in[i];
    switch (in_sizes[i]) {
      case 140:   if (c140 == 0) feat_mean = p; else feat_std = p; c140++; break;
      case 15360: embed = p; break;
      case 73216: early_w = p; break;
      case 65536: if (c65536 < 8) gin_w[c65536] = p; else late_w1 = p; c65536++; break;
      case 256:   if (c256 < 8) gin_b[c256] = p; c256++; break;
      case 32768: late_w2 = p; break;
      case 128:   pred_w = p; break;
      case 1:     pred_b = p; break;
      default: break;
    }
  }
  if (!feat_mean || !feat_std || !embed || !early_w || !late_w1 || !late_w2 ||
      !pred_w || !pred_b || c65536 < 9 || c256 < 8) return;

  const int M = CCFG * NN;  // 64000 rows per batch

  // Workspace layout
  size_t off = 0;
  auto take = [&](size_t b) { size_t o = off; off += (b + 255) & ~(size_t)255; return o; };
  size_t o_A    = take((size_t)M * 288 * 2);       // bf16 matmul input (reused, stride 288 or 256)
  size_t o_Y    = take((size_t)M * HIDD * 4);      // f32 GEMM output
  size_t o_H    = take((size_t)M * HIDD * 4);      // f32 activations
  size_t o_H1   = take((size_t)M * HIDD * 4);      // f32 activations (x1)
  size_t o_agg  = take((size_t)M * HIDD * 4);      // f32 edge aggregation
  size_t o_sum  = take((size_t)CCFG * HIDD * 4);
  size_t o_sq   = take((size_t)CCFG * HIDD * 4);
  size_t o_pool = take((size_t)BB * CCFG * HIDD * 4);
  size_t o_pkE  = take((size_t)9 * 16 * 512 * 2);
  size_t o_pkG  = take((size_t)8 * 8 * 16 * 512 * 2);
  size_t o_emb  = take((size_t)120 * EMBD * 2);
  size_t o_z1   = take((size_t)CCFG * HIDD * 4);
  size_t o_h1s  = take((size_t)CCFG * HIDD * 4);
  size_t o_z2   = take((size_t)CCFG * 128 * 4);
  size_t o_h2s  = take((size_t)CCFG * 128 * 4);
  size_t o_rt   = take((size_t)BB * CCFG * 4);
  if (off > ws_size) return;

  char* ws = (char*)d_ws;
  __hip_bfloat16* Abuf  = (__hip_bfloat16*)(ws + o_A);
  float* Ybuf  = (float*)(ws + o_Y);
  float* Hbuf  = (float*)(ws + o_H);
  float* H1buf = (float*)(ws + o_H1);
  float* Agg   = (float*)(ws + o_agg);
  float* SumB  = (float*)(ws + o_sum);
  float* SqB   = (float*)(ws + o_sq);
  float* Pool  = (float*)(ws + o_pool);
  __hip_bfloat16* PackE = (__hip_bfloat16*)(ws + o_pkE);
  __hip_bfloat16* PackG = (__hip_bfloat16*)(ws + o_pkG);
  __hip_bfloat16* EmbBf = (__hip_bfloat16*)(ws + o_emb);
  float* Z1  = (float*)(ws + o_z1);
  float* H1s = (float*)(ws + o_h1s);
  float* Z2  = (float*)(ws + o_z2);
  float* H2s = (float*)(ws + o_h2s);
  float* Rt  = (float*)(ws + o_rt);

  const size_t elemMH = (size_t)M * HIDD;
  const int gridMH = (int)((elemMH + 255) / 256);
  const dim3 gemmGrid(M / 128, HIDD / 64);

  // Weight prep (cheap, deterministic, per launch)
  pack_weight_bf16<<<(9 * 16 * 512 + 255) / 256, 256, 0, stream>>>(early_w, PackE, 286, 256, 9);
  for (int i = 0; i < 8; ++i)
    pack_weight_bf16<<<(8 * 16 * 512 + 255) / 256, 256, 0, stream>>>(
        gin_w[i], PackG + (size_t)i * 8 * 16 * 512, 256, 256, 8);
  renorm_embed<<<1, 128, 0, stream>>>(embed, EmbBf);
  hipMemsetAsync(Pool, 0, (size_t)BB * CCFG * HIDD * 4, stream);

  for (int b = 0; b < BB; ++b) {
    const float* nf   = node_feat + (size_t)b * NN * OPF;
    const int*   opc  = node_opcode + (size_t)b * NN;
    const float* cf   = node_config_feat + (size_t)b * CCFG * NCC * CFGF;
    const int*   cids = node_config_ids + (size_t)b * NCC;
    const int*   ei   = edge_index + (size_t)b * 2 * EE;
    float* pool_b = Pool + (size_t)b * CCFG * HIDD;

    // ---- early MLP ----
    build_early_A<<<(int)(((size_t)M * 288 + 255) / 256), 256, 0, stream>>>(
        nf, opc, EmbBf, feat_mean, feat_std, Abuf);
    scatter_cfg<<<(CCFG * NCC * CFGF + 255) / 256, 256, 0, stream>>>(cf, cids, Abuf);
    gemm_bf16_wmma<<<gemmGrid, 256, 0, stream>>>(Abuf, PackE, nullptr, Ybuf, M, 288, HIDD);
    hipMemsetAsync(SumB, 0, CCFG * HIDD * 4, stream);
    hipMemsetAsync(SqB,  0, CCFG * HIDD * 4, stream);
    col_stats<<<dim3(CCFG, 16), 256, 0, stream>>>(Ybuf, SumB, SqB, 16);
    norm_apply<<<gridMH, 256, 0, stream>>>(Ybuf, SumB, SqB, Hbuf, nullptr, /*gelu*/0);

    // ---- 4 GIN layers; layer 3 consumes x1 (reference quirk) ----
    for (int l = 0; l < 4; ++l) {
      const float* Hin = (l < 2) ? Hbuf : H1buf;     // l0:x l1:x0 l2:x1 l3:x1
      float* Hout = (l == 1) ? H1buf : Hbuf;         // x0,x1,x2,x3
      hipMemsetAsync(Agg, 0, elemMH * 4, stream);
      edge_agg<<<dim3(EE, CCFG), 256, 0, stream>>>(Hin, ei, Agg);
      combine_bf<<<gridMH, 256, 0, stream>>>(Hin, Agg, Abuf);

      gemm_bf16_wmma<<<gemmGrid, 256, 0, stream>>>(
          Abuf, PackG + (size_t)(2 * l) * 8 * 16 * 512, gin_b[2 * l], Ybuf, M, 256, HIDD);
      hipMemsetAsync(SumB, 0, CCFG * HIDD * 4, stream);
      hipMemsetAsync(SqB,  0, CCFG * HIDD * 4, stream);
      col_stats<<<dim3(CCFG, 16), 256, 0, stream>>>(Ybuf, SumB, SqB, 16);
      norm_apply<<<gridMH, 256, 0, stream>>>(Ybuf, SumB, SqB, nullptr, Abuf, /*elu*/1);

      gemm_bf16_wmma<<<gemmGrid, 256, 0, stream>>>(
          Abuf, PackG + (size_t)(2 * l + 1) * 8 * 16 * 512, gin_b[2 * l + 1], Ybuf, M, 256, HIDD);
      hipMemsetAsync(SumB, 0, CCFG * HIDD * 4, stream);
      hipMemsetAsync(SqB,  0, CCFG * HIDD * 4, stream);
      col_stats<<<dim3(CCFG, 16), 256, 0, stream>>>(Ybuf, SumB, SqB, 16);
      norm_apply<<<gridMH, 256, 0, stream>>>(Ybuf, SumB, SqB, Hout, nullptr, /*elu*/1);

      pool_acc<<<dim3(CCFG, 16), 256, 0, stream>>>(Hout, pool_b, 16);
    }

    // ---- late stage (tiny, plain f32 for accuracy) ----
    small_gemm<<<(CCFG * HIDD + 255) / 256, 256, 0, stream>>>(pool_b, late_w1, Z1, CCFG, 256, 256);
    normcfg_gelu<<<1, 256, 0, stream>>>(Z1, H1s, CCFG, 256);
    small_gemm<<<(CCFG * 128 + 255) / 256, 256, 0, stream>>>(H1s, late_w2, Z2, CCFG, 256, 128);
    normcfg_gelu<<<1, 128, 0, stream>>>(Z2, H2s, CCFG, 128);
    pred_k<<<1, 32, 0, stream>>>(H2s, pred_w, pred_b, Rt + b * CCFG, 128, CCFG);
  }

  loss_k<<<1, 256, 0, stream>>>(Rt, config_runtime, (float*)d_out);
}
